// MLP_44951127720297
// MI455X (gfx1250) — 8-bit matrix operand order repaired and checked by reading, compiled, not run on hardware
//
#include <hip/hip_runtime.h>
#include <math.h>

typedef __attribute__((ext_vector_type(16))) int   v16i;
typedef __attribute__((ext_vector_type(8)))  float v8f;
typedef __attribute__((ext_vector_type(4)))  int   v4i;
typedef __attribute__((ext_vector_type(2)))  int   v2i;

#define TM 128
#define TN 128
#define TK 128
#define LSTRIDE 144

#if __has_builtin(__builtin_amdgcn_global_load_async_to_lds_b128)
#define HAS_ASYNC_LDS 1
#else
#define HAS_ASYNC_LDS 0
#endif

__device__ __forceinline__ void async_cp16(const unsigned char* g, unsigned char* l) {
#if HAS_ASYNC_LDS
  __builtin_amdgcn_global_load_async_to_lds_b128(
      (__attribute__((address_space(1))) v4i*)g,
      (__attribute__((address_space(3))) v4i*)l, 0, 0);
#else
  *(v4i*)l = *(const v4i*)g;
#endif
}

template <int N>
__device__ __forceinline__ void wait_async() {
#if HAS_ASYNC_LDS
#if __has_builtin(__builtin_amdgcn_s_wait_asynccnt)
  __builtin_amdgcn_s_wait_asynccnt(N);
#else
  asm volatile("s_wait_asynccnt %0" ::"n"(N) : "memory");
#endif
#endif
}

__device__ __forceinline__ unsigned f32_to_e4m3(float x) {
  unsigned u    = __float_as_uint(x);
  unsigned sign = (u >> 24) & 0x80u;
  unsigned a    = u & 0x7FFFFFFFu;
  if (a >= 0x7F800000u) return sign | 0x7Fu;
  if (a >= 0x43E00000u) return sign | 0x7Eu;
  int exp       = (int)(a >> 23) - 127;
  unsigned mant = a & 0x7FFFFFu;
  if (exp < -6) {
    float f = __uint_as_float(a) * 512.0f;
    int m = (int)rintf(f);
    return sign | (unsigned)m;
  }
  unsigned m3   = mant >> 20;
  unsigned rest = mant & 0xFFFFFu;
  const unsigned half = 0x80000u;
  if (rest > half || (rest == half && (m3 & 1u))) m3++;
  if (m3 == 8u) { m3 = 0u; exp++; }
  return sign | ((unsigned)(exp + 7) << 3) | m3;
}

__device__ __forceinline__ float scale_from_absmax(float amax) {
  return fmaxf(amax * (1.0f / 448.0f), 1e-6f);
}

__global__ void mlp_init_scalars(float* p) {
  if (threadIdx.x < 8) p[threadIdx.x] = 0.0f;
}

__global__ __launch_bounds__(256) void mlp_absmax(const float* __restrict__ in,
                                                  long long n,
                                                  float* __restrict__ outAbs) {
  __shared__ float red[256];
  long long t      = (long long)blockIdx.x * 256 + threadIdx.x;
  long long stride = (long long)gridDim.x * 256 * 4;
  float m = 0.0f;
  for (long long i = t * 4; i + 3 < n; i += stride) {
    float4 v = *(const float4*)(in + i);
    m = fmaxf(m, fmaxf(fmaxf(fabsf(v.x), fabsf(v.y)),
                       fmaxf(fabsf(v.z), fabsf(v.w))));
  }
  red[threadIdx.x] = m;
  __syncthreads();
  for (int s = 128; s > 0; s >>= 1) {
    if ((int)threadIdx.x < s)
      red[threadIdx.x] = fmaxf(red[threadIdx.x], red[threadIdx.x + s]);
    __syncthreads();
  }
  if (threadIdx.x == 0)
    atomicMax((unsigned int*)outAbs, __float_as_uint(red[0]));
}

__global__ __launch_bounds__(256) void mlp_quant(const float* __restrict__ in,
                                                 unsigned char* __restrict__ out,
                                                 long long n,
                                                 const float* __restrict__ absPtr) {
  float inv = 1.0f / scale_from_absmax(*absPtr);
  long long t      = (long long)blockIdx.x * 256 + threadIdx.x;
  long long stride = (long long)gridDim.x * 256 * 4;
  for (long long i = t * 4; i + 3 < n; i += stride) {
    float4 v = *(const float4*)(in + i);
    unsigned p = f32_to_e4m3(v.x * inv)
               | (f32_to_e4m3(v.y * inv) << 8)
               | (f32_to_e4m3(v.z * inv) << 16)
               | (f32_to_e4m3(v.w * inv) << 24);
    *(unsigned*)(out + i) = p;
  }
}

__device__ __forceinline__ v16i load_bfrag(const unsigned char* lB, int j,
                                           int lane, int hiB) {
  v16i b;
  const unsigned char* base = lB + (j * 16 + (lane & 15)) * LSTRIDE;
#pragma unroll
  for (int q = 0; q < 4; ++q) {
    v4i d = *(const v4i*)(base + q * 32 + hiB);
    b[4 * q]     = d.x;
    b[4 * q + 1] = d.y;
    b[4 * q + 2] = d.z;
    b[4 * q + 3] = d.w;
  }
  return b;
}

__device__ __forceinline__ void stage_tiles(const unsigned char* __restrict__ A8,
                                            const unsigned char* __restrict__ B8,
                                            unsigned char* lA, unsigned char* lB,
                                            int bm, int bn, int K, int k0,
                                            int ldRow, int ldCol) {
#pragma unroll
  for (int i = 0; i < 4; ++i) {
    int r = ldRow + i * 32;
    async_cp16(A8 + (size_t)(bm + r) * K + k0 + ldCol, lA + r * LSTRIDE + ldCol);
    async_cp16(B8 + (size_t)(bn + r) * K + k0 + ldCol, lB + r * LSTRIDE + ldCol);
  }
}

__global__ __launch_bounds__(256) void mlp_gemm_fp8(
    const unsigned char* __restrict__ A8,
    const unsigned char* __restrict__ B8,
    float* __restrict__ C,
    int M, int N, int K,
    const float* __restrict__ absA,
    const float* __restrict__ absB,
    float* __restrict__ hAbsOut,
    int doRelu2) {
  __shared__ unsigned char lA[2][TM * LSTRIDE];
  __shared__ unsigned char lB[2][TN * LSTRIDE];

  const int tid  = threadIdx.x;
  const int lane = tid & 31;
  const int wave = tid >> 5;
  const int bm   = blockIdx.x * TM;
  const int bn   = blockIdx.y * TN;

  const float outScale = scale_from_absmax(*absA) * scale_from_absmax(*absB);

  v8f acc[8] = {};

  const int ldRow = tid >> 3;
  const int ldCol = (tid & 7) * 16;

  const int rowA = wave * 16 + (lane & 15);
  const int hiA  = (lane >> 4) * 16;
  const int hiB  = (lane >> 4) * 16;

  stage_tiles(A8, B8, lA[0], lB[0], bm, bn, K, 0, ldRow, ldCol);

  int buf = 0;
  for (int k0 = 0; k0 < K; k0 += TK) {
    if (k0 + TK < K) {
      stage_tiles(A8, B8, lA[buf ^ 1], lB[buf ^ 1], bm, bn, K, k0 + TK,
                  ldRow, ldCol);
      wait_async<8>();
    } else {
      wait_async<0>();
    }
    __syncthreads();

    v16i afrag;
    {
      const unsigned char* base = &lA[buf][rowA * LSTRIDE];
#pragma unroll
      for (int p = 0; p < 8; ++p) {
        v2i d = *(const v2i*)(base + (p >> 1) * 32 + (p & 1) * 8 + hiA);
        afrag[2 * p]     = d.x;
        afrag[2 * p + 1] = d.y;
      }
    }

    v16i bfr0 = load_bfrag(lB[buf], 0, lane, hiB);
    v16i bfr1;
#pragma unroll
    for (int j = 0; j < 8; ++j) {
      if ((j & 1) == 0) {
        if (j < 7) bfr1 = load_bfrag(lB[buf], j + 1, lane, hiB);
        acc[j] = __builtin_amdgcn_wmma_f32_16x16x128_fp8_fp8(
            afrag, bfr0, (short)0, acc[j], false, false);
      } else {
        if (j < 7) bfr0 = load_bfrag(lB[buf], j + 1, lane, hiB);
        acc[j] = __builtin_amdgcn_wmma_f32_16x16x128_fp8_fp8(
            afrag, bfr1, (short)0, acc[j], false, false);
      }
    }
    __syncthreads();
    buf ^= 1;
  }

  float localMax = 0.0f;
#pragma unroll
  for (int j = 0; j < 8; ++j) {
    int n = bn + j * 16 + (lane & 15);
#pragma unroll
    for (int r = 0; r < 8; ++r) {
      int m = bm + wave * 16 + r + (lane >> 4) * 8;
      float v = acc[j][r] * outScale;
      if (doRelu2) {
        v = v > 0.0f ? v * v : 0.0f;
        localMax = fmaxf(localMax, v);
      }
      C[(size_t)m * N + n] = v;
    }
  }

  if (hAbsOut != nullptr) {
    float* red = (float*)&lA[0][0];
    red[tid] = localMax;
    __syncthreads();
    for (int s = 128; s > 0; s >>= 1) {
      if (tid < s) red[tid] = fmaxf(red[tid], red[tid + s]);
      __syncthreads();
    }
    if (tid == 0) atomicMax((unsigned int*)hAbsOut, __float_as_uint(red[0]));
  }
}

extern "C" void kernel_launch(void* const* d_in, const int* in_sizes, int n_in,
                              void* d_out, int out_size, void* d_ws, size_t ws_size,
                              hipStream_t stream) {
  const float* x      = (const float*)d_in[0];
  const float* w_fc   = (const float*)d_in[1];
  const float* w_proj = (const float*)d_in[2];
  float* out          = (float*)d_out;

  const int dim   = (int)(sqrt((double)in_sizes[1] / 4.0) + 0.5);
  const int hid   = 4 * dim;
  const int Nrows = (int)((long long)in_sizes[0] / dim);

  const long long nX  = (long long)Nrows * dim;
  const long long nWf = (long long)hid * dim;
  const long long nWp = (long long)dim * hid;
  const long long nH  = (long long)Nrows * hid;

  char* w = (char*)d_ws;
  float* scal = (float*)w;
  unsigned char* x8  = (unsigned char*)(w + 256);
  unsigned char* wf8 = x8 + nX;
  unsigned char* wp8 = wf8 + nWf;
  float*         h   = (float*)(wp8 + nWp);
  unsigned char* h8  = (unsigned char*)(h + nH);

  mlp_init_scalars<<<1, 32, 0, stream>>>(scal);

  mlp_absmax<<<1024, 256, 0, stream>>>(x,      nX,  scal + 0);
  mlp_absmax<<<1024, 256, 0, stream>>>(w_fc,   nWf, scal + 1);
  mlp_absmax<<<1024, 256, 0, stream>>>(w_proj, nWp, scal + 2);

  mlp_quant<<<1024, 256, 0, stream>>>(x,      x8,  nX,  scal + 0);
  mlp_quant<<<1024, 256, 0, stream>>>(w_fc,   wf8, nWf, scal + 1);
  mlp_quant<<<1024, 256, 0, stream>>>(w_proj, wp8, nWp, scal + 2);

  dim3 g1(Nrows / TM, hid / TN);
  mlp_gemm_fp8<<<g1, 256, 0, stream>>>(x8, wf8, h, Nrows, hid, dim,
                                       scal + 0, scal + 1, scal + 3, 1);

  mlp_quant<<<2048, 256, 0, stream>>>(h, h8, nH, scal + 3);

  dim3 g2(Nrows / TM, dim / TN);
  mlp_gemm_fp8<<<g2, 256, 0, stream>>>(h8, wp8, out, Nrows, dim, hid,
                                       scal + 3, scal + 2, nullptr, 0);
}
